// SCHetNet_48610439856560
// MI455X (gfx1250) — compile-verified, matrix operand order checked by reading, not run on hardware
//
#include <hip/hip_runtime.h>

typedef __attribute__((ext_vector_type(16))) _Float16 v16h;
typedef __attribute__((ext_vector_type(8)))  _Float16 v8h;
typedef __attribute__((ext_vector_type(8)))  float    v8f;

#define NN 193536      // B*(S+R)
#define EE 3072000
#define BB 512
#define SS 38
#define RR 340
#define PP 378         // S+R
#define HCN 128
#define NK (NN*HCN)    // 24,772,608

// Non-returning hardware float atomic (GLOBAL_ATOMIC_ADD_F32, tracked on STOREcnt)
__device__ __forceinline__ void gatomic_add_f32(float* p, float v) {
    asm volatile("global_atomic_add_f32 %0, %1, off" :: "v"(p), "v"(v) : "memory");
}

// ---------------- BN statistics ----------------
__global__ void zero_stats(float* st) {
    if (threadIdx.x < 32) st[threadIdx.x] = 0.0f;
}

__global__ void bn_stats_s(const float* __restrict__ s_x, float* __restrict__ st) {
    int tid = blockIdx.x * blockDim.x + threadIdx.x;
    int stride = gridDim.x * blockDim.x;
    float s[4] = {0,0,0,0}, q[4] = {0,0,0,0};
    for (int r = tid; r < BB*SS; r += stride) {
        #pragma unroll
        for (int c = 0; c < 4; ++c) { float v = s_x[r*4+c]; s[c] += v; q[c] += v*v; }
    }
    #pragma unroll
    for (int c = 0; c < 4; ++c) { gatomic_add_f32(&st[c], s[c]); gatomic_add_f32(&st[4+c], q[c]); }
}

__global__ void bn_stats_r(const float* __restrict__ r_x, float* __restrict__ st) {
    int tid = blockIdx.x * blockDim.x + threadIdx.x;
    int stride = gridDim.x * blockDim.x;
    float s[3] = {0,0,0}, q[3] = {0,0,0};
    for (int r = tid; r < BB*RR; r += stride) {
        #pragma unroll
        for (int c = 0; c < 3; ++c) { float v = r_x[(size_t)r*8+c]; s[c] += v; q[c] += v*v; }
    }
    #pragma unroll
    for (int c = 0; c < 3; ++c) { gatomic_add_f32(&st[8+c], s[c]); gatomic_add_f32(&st[11+c], q[c]); }
}

__global__ void bn_finalize(const float* __restrict__ bn_s_w, const float* __restrict__ bn_s_b,
                            const float* __restrict__ bn_r_w, const float* __restrict__ bn_r_b,
                            float* __restrict__ st) {
    int t = threadIdx.x;
    if (t < 4) {
        float m = st[t] / (float)(BB*SS);
        float v = st[4+t] / (float)(BB*SS) - m*m;
        float a = bn_s_w[t] * rsqrtf(v + 1e-5f);
        st[16+t] = a;
        st[20+t] = bn_s_b[t] - m*a;
    } else if (t >= 8 && t < 11) {
        int c = t - 8;
        float m = st[8+c] / (float)(BB*RR);
        float v = st[11+c] / (float)(BB*RR) - m*m;
        float a = bn_r_w[c] * rsqrtf(v + 1e-5f);
        st[24+c] = a;
        st[27+c] = bn_r_b[c] - m*a;
    }
}

// ---------------- embed: BN affine + linear + leaky, concat layout ----------------
__global__ void embed_kernel(const float* __restrict__ s_x, const float* __restrict__ r_x,
                             const float* __restrict__ lin_s_w, const float* __restrict__ lin_r_w,
                             const float* __restrict__ st,
                             float* __restrict__ x, _Float16* __restrict__ xh) {
    size_t gid = (size_t)blockIdx.x * 256 + threadIdx.x;
    if (gid >= (size_t)NK) return;
    int i = (int)(gid >> 7);
    int k = (int)(gid & 127);
    int b = i / PP, p = i % PP;
    float acc = 0.0f;
    if (p < SS) {
        int row = b*SS + p;
        #pragma unroll
        for (int c = 0; c < 4; ++c) {
            float f = s_x[row*4+c] * st[16+c] + st[20+c];
            acc += f * lin_s_w[c*HCN + k];
        }
    } else {
        int row = b*RR + (p - SS);
        #pragma unroll
        for (int c = 0; c < 3; ++c) {
            float f = r_x[(size_t)row*8+c] * st[24+c] + st[27+c];
            acc += f * lin_r_w[c*HCN + k];
        }
    }
    acc = acc > 0.0f ? acc : 0.2f*acc;
    x[gid] = acc;
    xh[gid] = (_Float16)acc;
}

__global__ void ox_extract(const float* __restrict__ r_x, float* __restrict__ o_x) {
    int t = blockIdx.x * blockDim.x + threadIdx.x;
    if (t < BB*5) {
        int b = t / 5, j = t % 5;
        o_x[t] = r_x[(size_t)b*RR*8 + 3 + j];
    }
}

// ---------------- degree / norm ----------------
__global__ void deg_init(float* __restrict__ deg) {
    int i = blockIdx.x * blockDim.x + threadIdx.x;
    if (i < NN) deg[i] = 1.0f;   // self loop
}
__global__ void deg_acc(const int* __restrict__ ei, float* __restrict__ deg) {
    int e = blockIdx.x * blockDim.x + threadIdx.x;
    if (e < EE) gatomic_add_f32(&deg[ei[(size_t)EE + e]], 1.0f);
}
__global__ void dinv_kernel(float* __restrict__ deg) {
    int i = blockIdx.x * blockDim.x + threadIdx.x;
    if (i < NN) deg[i] = rsqrtf(deg[i]);
}
__global__ void norm_kernel(const int* __restrict__ ei, const float* __restrict__ dinv,
                            float* __restrict__ nrm) {
    int e = blockIdx.x * blockDim.x + threadIdx.x;
    if (e < EE) nrm[e] = dinv[ei[e]] * dinv[ei[(size_t)EE + e]];
}

// ---------------- pack conv_w into WMMA B-fragment order (f16) ----------------
// Bp layout per layer: [(kt*8+nt)*32 + lane][16 halves]
__global__ void pack_w(const float* __restrict__ conv_w, _Float16* __restrict__ Bp) {
    int t = blockIdx.x * blockDim.x + threadIdx.x;
    if (t >= 4*4*8*32*16) return;
    int i    =  t        & 15;
    int lane = (t >> 4)  & 31;
    int nt   = (t >> 9)  & 7;
    int kt   = (t >> 12) & 3;
    int l    =  t >> 14;
    int n = nt*16 + (lane & 15);
    int k = kt*32 + ((lane >> 4) << 3) + (i < 8 ? i : i + 8);
    Bp[t] = (_Float16)conv_w[(size_t)l*HCN*HCN + (size_t)k*HCN + n];
}

// ---------------- WMMA GEMM: xw = x_h (N x 128) @ W (128 x 128) ----------------
// B fragments staged once per block into LDS (32 KB of 320 KB WGP LDS),
// shared by all 8 waves -> B reuse moves from L2 to the DS path.
__global__ void gemm_xw(const _Float16* __restrict__ xh, const _Float16* __restrict__ Bp,
                        float* __restrict__ xw) {
    __shared__ _Float16 lB[HCN * HCN];   // 16384 halves = 32 KB

    // cooperative copy: 256 threads x 8 uint4 (16B) = 32 KB
    {
        const uint4* src = (const uint4*)Bp;
        uint4* dst = (uint4*)lB;
        #pragma unroll
        for (int j = 0; j < 8; ++j)
            dst[threadIdx.x + j*256] = src[threadIdx.x + j*256];
    }
    __syncthreads();

    int wave = blockIdx.x * 8 + (threadIdx.x >> 5);
    int lane = threadIdx.x & 31;
    int row0 = wave * 16;
    if (row0 >= NN) return;
    int r    = lane & 15;
    int koff = (lane >> 4) * 8;
    const _Float16* arow = xh + (size_t)(row0 + r) * HCN + koff;

    v8f acc[8];
    v8f zero = {0,0,0,0,0,0,0,0};
    #pragma unroll
    for (int nt = 0; nt < 8; ++nt) acc[nt] = zero;

    #pragma unroll
    for (int kt = 0; kt < 4; ++kt) {
        v8h lo = *(const v8h*)(arow + kt*32);
        v8h hi = *(const v8h*)(arow + kt*32 + 16);
        v16h a = __builtin_shufflevector(lo, hi, 0,1,2,3,4,5,6,7,8,9,10,11,12,13,14,15);
        #pragma unroll
        for (int nt = 0; nt < 8; ++nt) {
            v16h b = *(const v16h*)(lB + (((kt*8 + nt)*32 + lane) * 16));
            acc[nt] = __builtin_amdgcn_wmma_f32_16x16x32_f16(
                false, a, false, b, (short)0, acc[nt], false, false);
        }
    }
    int mrow = (lane < 16) ? 0 : 8;
    int ncol = lane & 15;
    #pragma unroll
    for (int nt = 0; nt < 8; ++nt) {
        #pragma unroll
        for (int j = 0; j < 8; ++j) {
            xw[(size_t)(row0 + mrow + j) * HCN + nt*16 + ncol] = acc[nt][j];
        }
    }
}

// ---------------- agg = self-loop + edge scatter ----------------
__global__ void self_init(const float* __restrict__ dinv, const float* __restrict__ xw,
                          float* __restrict__ agg) {
    size_t gid = (size_t)blockIdx.x * 256 + threadIdx.x;
    if (gid >= (size_t)NK) return;
    int i = (int)(gid >> 7);
    float d = dinv[i];
    agg[gid] = d * d * xw[gid];
}

__global__ void scatter_kernel(const int* __restrict__ ei, const float* __restrict__ nrm,
                               const float* __restrict__ xw, float* __restrict__ agg) {
    int e = blockIdx.x * 8 + (threadIdx.x >> 5);
    if (e >= EE) return;
    int lane = threadIdx.x & 31;
    int src = ei[e];
    int dst = ei[(size_t)EE + e];
    float w = nrm[e];
    const float4* vp = (const float4*)(xw + (size_t)src*HCN) + lane;
    float* ap = agg + (size_t)dst*HCN + lane*4;
    __builtin_prefetch(ap, 0, 1);           // warm destination line (global_prefetch_b8)
    float4 v = *vp;
    gatomic_add_f32(ap + 0, w * v.x);
    gatomic_add_f32(ap + 1, w * v.y);
    gatomic_add_f32(ap + 2, w * v.z);
    gatomic_add_f32(ap + 3, w * v.w);
}

__global__ void update_kernel(const float* __restrict__ agg, const float* __restrict__ cb,
                              float* __restrict__ x, _Float16* __restrict__ xh) {
    size_t gid = (size_t)blockIdx.x * 256 + threadIdx.x;
    if (gid >= (size_t)NK) return;
    int k = (int)(gid & 127);
    float a = agg[gid] + cb[k];
    a = a > 0.0f ? a : 0.2f*a;
    float nx = x[gid] + a;
    x[gid] = nx;
    xh[gid] = (_Float16)nx;
}

// ---------------- pooling: mean over HC per node ----------------
__global__ void pool_kernel(const float* __restrict__ x, float* __restrict__ pooled) {
    int node = blockIdx.x * 8 + (threadIdx.x >> 5);
    if (node >= NN) return;
    int lane = threadIdx.x & 31;
    const float4* vp = (const float4*)(x + (size_t)node*HCN) + lane;
    float4 v = *vp;
    float s = v.x + v.y + v.z + v.w;
    #pragma unroll
    for (int o = 16; o > 0; o >>= 1) s += __shfl_xor(s, o, 32);
    if (lane == 0) pooled[node] = s * (1.0f/HCN);
}

// ---------------- head ----------------
__global__ void head_kernel(const float* __restrict__ pooled,
                            const float* __restrict__ linr_w, const float* __restrict__ linr_b,
                            const float* __restrict__ o_x,
                            const float* __restrict__ w1, const float* __restrict__ b1,
                            const float* __restrict__ w2, const float* __restrict__ b2,
                            float* __restrict__ out) {
    int b = blockIdx.x * blockDim.x + threadIdx.x;
    if (b >= BB) return;
    float z[7];
    #pragma unroll
    for (int j = 0; j < 7; ++j) z[j] = linr_b[j];
    for (int p = 0; p < PP; ++p) {
        float pv = pooled[b*PP + p];
        #pragma unroll
        for (int j = 0; j < 7; ++j) z[j] += pv * linr_w[p*7 + j];
    }
    float zs = 0.0f;
    #pragma unroll
    for (int j = 0; j < 7; ++j) { z[j] = __expf(z[j]); zs += z[j]; }
    float inv = 1.0f / (zs + 1.0f);
    float o[7];
    #pragma unroll
    for (int j = 0; j < 7; ++j) o[j] = b2[j];
    float f[5];
    #pragma unroll
    for (int c = 0; c < 5; ++c) f[c] = o_x[b*5 + c];
    for (int k = 0; k < HCN; ++k) {
        float t = b1[k];
        #pragma unroll
        for (int c = 0; c < 5; ++c) t += f[c] * w1[c*HCN + k];
        t = t > 0.0f ? t : 0.2f*t;
        #pragma unroll
        for (int j = 0; j < 7; ++j) o[j] += t * w2[k*7 + j];
    }
    #pragma unroll
    for (int j = 0; j < 7; ++j) out[b*7 + j] = z[j] * inv * __expf(o[j]);
}

extern "C" void kernel_launch(void* const* d_in, const int* in_sizes, int n_in,
                              void* d_out, int out_size, void* d_ws, size_t ws_size,
                              hipStream_t stream) {
    const float* s_x     = (const float*)d_in[0];
    const float* r_x     = (const float*)d_in[1];
    const int*   ei      = (const int*)  d_in[2];
    const float* bn_s_w  = (const float*)d_in[3];
    const float* bn_s_b  = (const float*)d_in[4];
    const float* bn_r_w  = (const float*)d_in[5];
    const float* bn_r_b  = (const float*)d_in[6];
    const float* lin_s_w = (const float*)d_in[7];
    const float* lin_r_w = (const float*)d_in[8];
    const float* conv_w  = (const float*)d_in[9];
    const float* conv_b  = (const float*)d_in[10];
    const float* linr_w  = (const float*)d_in[11];
    const float* linr_b  = (const float*)d_in[12];
    const float* lino_w1 = (const float*)d_in[13];
    const float* lino_b1 = (const float*)d_in[14];
    const float* lino_w2 = (const float*)d_in[15];
    const float* lino_b2 = (const float*)d_in[16];

    char* wsb = (char*)d_ws;
    size_t off = 0;
    auto alloc = [&](size_t bytes) -> void* {
        void* p = wsb + off;
        off = (off + bytes + 255) & ~(size_t)255;
        return p;
    };
    float*     x      = (float*)    alloc((size_t)NK * 4);
    _Float16*  xh     = (_Float16*) alloc((size_t)NK * 2);
    float*     xw     = (float*)    alloc((size_t)NK * 4);
    float*     agg    = (float*)    alloc((size_t)NK * 4);
    float*     dinv   = (float*)    alloc((size_t)NN * 4);
    float*     nrm    = (float*)    alloc((size_t)EE * 4);
    float*     pooled = (float*)    alloc((size_t)NN * 4);
    float*     o_x    = (float*)    alloc((size_t)BB * 5 * 4);
    _Float16*  Bp     = (_Float16*) alloc((size_t)4 * HCN * HCN * 2);
    float*     st     = (float*)    alloc(32 * 4);

    // BN stats + affine
    zero_stats<<<1, 32, 0, stream>>>(st);
    bn_stats_s<<<64, 256, 0, stream>>>(s_x, st);
    bn_stats_r<<<64, 256, 0, stream>>>(r_x, st);
    bn_finalize<<<1, 32, 0, stream>>>(bn_s_w, bn_s_b, bn_r_w, bn_r_b, st);

    // embed + o_x + weight packing
    embed_kernel<<<NK/256, 256, 0, stream>>>(s_x, r_x, lin_s_w, lin_r_w, st, x, xh);
    ox_extract<<<10, 256, 0, stream>>>(r_x, o_x);
    pack_w<<<256, 256, 0, stream>>>(conv_w, Bp);

    // degree / norm
    deg_init<<<NN/256, 256, 0, stream>>>(dinv);
    deg_acc<<<EE/256, 256, 0, stream>>>(ei, dinv);
    dinv_kernel<<<NN/256, 256, 0, stream>>>(dinv);
    norm_kernel<<<EE/256, 256, 0, stream>>>(ei, dinv, nrm);

    // GCN layers
    for (int l = 0; l < 4; ++l) {
        gemm_xw<<<NN/16/8, 256, 0, stream>>>(xh, Bp + (size_t)l*HCN*HCN, xw);
        self_init<<<NK/256, 256, 0, stream>>>(dinv, xw, agg);
        scatter_kernel<<<EE/8, 256, 0, stream>>>(ei, nrm, xw, agg);
        update_kernel<<<NK/256, 256, 0, stream>>>(agg, conv_b + l*HCN, x, xh);
    }

    // pooling + head
    pool_kernel<<<NN/8, 256, 0, stream>>>(x, pooled);
    head_kernel<<<2, 256, 0, stream>>>(pooled, linr_w, linr_b, o_x,
                                       lino_w1, lino_b1, lino_w2, lino_b2,
                                       (float*)d_out);
}
